// Edge_Cycle_50869592655481
// MI455X (gfx1250) — matrix operand order repaired and checked by reading, not compiled, not run on hardware
//
#include <hip/hip_runtime.h>

typedef __bf16 bf16_t;
typedef __attribute__((ext_vector_type(16))) __bf16 v16bf;
typedef __attribute__((ext_vector_type(8)))  __bf16 v8bf;
typedef __attribute__((ext_vector_type(8)))  float  v8f;

static constexpr int kH      = 128;
static constexpr int kNodes  = 60000;
static constexpr int kEdges  = 100000;
static constexpr int kC5     = 30000;
static constexpr int kC6     = 30000;
static constexpr int kERows  = 2 * kEdges;    // 200000
static constexpr int kC5Rows = 5 * kC5;       // 150000
static constexpr int kC6Rows = 6 * kC6;       // 180000

// ----------------------------- utility kernels -----------------------------

__global__ void zero_f32(float* __restrict__ p, size_t n) {
    size_t i  = (size_t)blockIdx.x * blockDim.x + threadIdx.x;
    size_t st = (size_t)gridDim.x * blockDim.x;
    for (; i < n; i += st) p[i] = 0.0f;
}

// W [K,N] f32 row-major  ->  Wt [N,K] bf16 row-major (B-fragment friendly)
__global__ void transpose_to_bf16(const float* __restrict__ W, bf16_t* __restrict__ Wt,
                                  int K, int N) {
    size_t total = (size_t)K * N;
    size_t i  = (size_t)blockIdx.x * blockDim.x + threadIdx.x;
    size_t st = (size_t)gridDim.x * blockDim.x;
    for (; i < total; i += st) {
        int k = (int)(i / N), n = (int)(i % N);
        Wt[(size_t)n * K + k] = (bf16_t)W[i];
    }
}

// segment_sum(src[rows,128], atoms) -> dst[nodes,128]   (one block per row)
__global__ __launch_bounds__(kH) void scatter_rows(const float* __restrict__ src,
                                                   const int* __restrict__ atoms,
                                                   float* __restrict__ dst) {
    int row = blockIdx.x, ch = threadIdx.x;
    atomicAdd(&dst[(size_t)atoms[row] * kH + ch], src[(size_t)row * kH + ch]);
}

// First _gather1 + source side of second _gather1:
//   m0_r = ns1[atom_r]; m1 = sum_r m0_r; ns2[atom_r] += concat[m0_r, m1]
template<int P>
__global__ __launch_bounds__(kH) void cyc_gather_scatter(const float* __restrict__ ns1,
                                                         const int* __restrict__ atoms,
                                                         float* __restrict__ ns2) {
    int sg = blockIdx.x, ch = threadIdx.x, base = sg * P;
    int a[P]; float v[P]; float s = 0.f;
#pragma unroll
    for (int r = 0; r < P; ++r) { a[r] = atoms[base + r]; v[r] = ns1[(size_t)a[r] * kH + ch]; s += v[r]; }
#pragma unroll
    for (int r = 0; r < P; ++r) {
        atomicAdd(&ns2[(size_t)a[r] * 256 + ch],       v[r]);
        atomicAdd(&ns2[(size_t)a[r] * 256 + kH + ch],  s);
    }
}

// Second _gather1 destination: e2c[row] = concat[ ns2[atom_r], sum_r ns2[atom_r] ]  (bf16)
template<int P>
__global__ __launch_bounds__(256) void prep_e2c(const float* __restrict__ ns2,
                                                const int* __restrict__ atoms,
                                                bf16_t* __restrict__ A) {
    int sg = blockIdx.x, ch = threadIdx.x, base = sg * P;
    float v[P]; float s = 0.f;
#pragma unroll
    for (int r = 0; r < P; ++r) { v[r] = ns2[(size_t)atoms[base + r] * 256 + ch]; s += v[r]; }
#pragma unroll
    for (int r = 0; r < P; ++r) {
        size_t ro = (size_t)(base + r) * 512;
        A[ro + ch]       = (bf16_t)v[r];
        A[ro + 256 + ch] = (bf16_t)s;
    }
}

// ns3[atom] += concat[e2c_out(128), crep(128)]
__global__ __launch_bounds__(kH) void scatter_cat(const float* __restrict__ e2co,
                                                  const float* __restrict__ crep,
                                                  const int* __restrict__ atoms,
                                                  float* __restrict__ ns3) {
    int row = blockIdx.x, ch = threadIdx.x;
    int a = atoms[row];
    atomicAdd(&ns3[(size_t)a * 256 + ch],      e2co[(size_t)row * kH + ch]);
    atomicAdd(&ns3[(size_t)a * 256 + kH + ch], crep[(size_t)row * kH + ch]);
}

// edge-side gather: rows 2e,2e+1 get concat[ns3[a], ns3[a0]+ns3[a1]] (bf16)
__global__ __launch_bounds__(256) void prep_edgeA(const float* __restrict__ ns3,
                                                  const int* __restrict__ eatoms,
                                                  bf16_t* __restrict__ A) {
    int e = blockIdx.x, ch = threadIdx.x;
    int a0 = eatoms[2 * e], a1 = eatoms[2 * e + 1];
    float v0 = ns3[(size_t)a0 * 256 + ch];
    float v1 = ns3[(size_t)a1 * 256 + ch];
    float s  = v0 + v1;
    size_t r0 = (size_t)(2 * e) * 512, r1 = r0 + 512;
    A[r0 + ch] = (bf16_t)v0;  A[r0 + 256 + ch] = (bf16_t)s;
    A[r1 + ch] = (bf16_t)v1;  A[r1 + 256 + ch] = (bf16_t)s;
}

// out = bf16(a + (1+*eps) * b)
__global__ void add_scaled_bf16(const float* __restrict__ a, const float* __restrict__ b,
                                const float* __restrict__ eps, bf16_t* __restrict__ out,
                                size_t n) {
    float sc = 1.0f + eps[0];
    size_t i  = (size_t)blockIdx.x * blockDim.x + threadIdx.x;
    size_t st = (size_t)gridDim.x * blockDim.x;
    for (; i < n; i += st) out[i] = (bf16_t)(a[i] + sc * b[i]);
}

// y = relu((x-mean)*rsqrt(var+eps)*gamma + beta), bf16 -> bf16 (hidden layer)
__global__ void bn_apply_b2b(const bf16_t* __restrict__ x, const float* __restrict__ stats,
                             const float* __restrict__ gamma, const float* __restrict__ beta,
                             float invM, bf16_t* __restrict__ out, int M, int C) {
    int ch = threadIdx.x;
    float mean = stats[ch] * invM;
    float var  = stats[C + ch] * invM - mean * mean;
    float sc = rsqrtf(var + 1e-5f) * gamma[ch];
    float sh = beta[ch] - mean * sc;
    int r0 = blockIdx.x * 16;
    int r1 = r0 + 16 < M ? r0 + 16 : M;
    for (int r = r0; r < r1; ++r) {
        size_t o = (size_t)r * C + ch;
        float v = (float)x[o] * sc + sh;
        out[o] = (bf16_t)(v > 0.f ? v : 0.f);
    }
}

// y = relu(bn(x)) f32 in-place (final layer, output buffer)
__global__ void bn_apply_f32(float* __restrict__ x, const float* __restrict__ stats,
                             const float* __restrict__ gamma, const float* __restrict__ beta,
                             float invM, int M, int C) {
    int ch = threadIdx.x;
    float mean = stats[ch] * invM;
    float var  = stats[C + ch] * invM - mean * mean;
    float sc = rsqrtf(var + 1e-5f) * gamma[ch];
    float sh = beta[ch] - mean * sc;
    int r0 = blockIdx.x * 16;
    int r1 = r0 + 16 < M ? r0 + 16 : M;
    for (int r = r0; r < r1; ++r) {
        size_t o = (size_t)r * C + ch;
        float v = x[o] * sc + sh;
        x[o] = v > 0.f ? v : 0.f;
    }
}

// ----------------------------- WMMA GEMM -----------------------------------
// C[M,N] = A[M,K](bf16 row-major) @ Wt[N,K](bf16 row-major, i.e. W^T)
// + optional bias[n] + optional (1+*res_eps)*res[M,N].
// Block = 256 threads = 8 waves; wave owns 16 rows x NT*16 cols (NT independent
// 16x16 f32 accumulators). grid.y = N/(NT*16). If STATS, per-column sum/sumsq of
// the f32 result are reduced via LDS and atomically added to stats[0..N)/[N..2N).
// M % 16 == 0, K % 32 == 0, N % (NT*16) == 0.
template<int NT, bool STATS>
__global__ __launch_bounds__(256)
void gemm_wmma_bf16(const bf16_t* __restrict__ A, const bf16_t* __restrict__ Wt,
                    const float* __restrict__ bias,
                    const float* __restrict__ res, const float* __restrict__ res_eps,
                    float* __restrict__ outF, bf16_t* __restrict__ outH,
                    float* __restrict__ stats,
                    int M, int K, int N) {
    __shared__ float sstat[NT * 32];            // [NT*16 sums][NT*16 sumsqs]
    const int wave  = threadIdx.x >> 5;
    const int lane  = threadIdx.x & 31;
    const int mbase = (blockIdx.x * 8 + wave) * 16;
    const int nbase = blockIdx.y * (NT * 16);
    const int l15   = lane & 15;
    const int half  = lane >> 4;
    const bool active = (mbase < M);

    if (STATS) {
        for (int i = threadIdx.x; i < NT * 32; i += 256) sstat[i] = 0.f;
        __syncthreads();
    } else if (!active) {
        return;                                 // wave-uniform, no barriers in this path
    }

    if (active) {
        v8f acc[NT] = {};
        // A layout (16x32 bf16): lane m=l15, elem e -> k=16*half+e
        const bf16_t* arow = A + (size_t)(mbase + l15) * K;
        // B layout (32x16 bf16): lane n=l15, elem e -> k=16*half+e (contiguous in Wt rows)
        const bf16_t* b0 = Wt + (size_t)(nbase + l15) * K + 16 * half;
        const size_t bstride = (size_t)16 * K;

        for (int k = 0; k < K; k += 32) {
            v16bf a = *(const v16bf*)(arow + k + 16 * half);
#pragma unroll
            for (int j = 0; j < NT; ++j) {
                v16bf bv = *(const v16bf*)(b0 + j * bstride + k);
                acc[j] = __builtin_amdgcn_wmma_f32_16x16x32_bf16(false, a, false, bv,
                                                                 (short)0, acc[j], false, false);
            }
        }

        float rs = 0.0f;
        if (res) rs = 1.0f + res_eps[0];
#pragma unroll
        for (int j = 0; j < NT; ++j) {
            int n = nbase + j * 16 + l15;
            float bb = bias ? bias[n] : 0.0f;
            float s = 0.f, s2 = 0.f;
#pragma unroll
            for (int r = 0; r < 8; ++r) {       // C/D layout: VGPR r -> row 8*half + r
                int m = mbase + 8 * half + r;
                size_t o = (size_t)m * N + n;
                float v = acc[j][r] + bb;
                if (res) v += rs * res[o];
                if (outF) outF[o] = v;
                if (outH) outH[o] = (bf16_t)v;
                s += v; s2 += v * v;
            }
            if (STATS) {
                atomicAdd(&sstat[j * 16 + l15], s);            // ds_add_f32
                atomicAdd(&sstat[NT * 16 + j * 16 + l15], s2);
            }
        }
    }

    if (STATS) {
        __syncthreads();
        for (int i = threadIdx.x; i < NT * 16; i += 256) {
            atomicAdd(&stats[nbase + i],     sstat[i]);
            atomicAdd(&stats[N + nbase + i], sstat[NT * 16 + i]);
        }
    }
}

// ----------------------------- host driver ---------------------------------

extern "C" void kernel_launch(void* const* d_in, const int* in_sizes, int n_in,
                              void* d_out, int out_size, void* d_ws, size_t ws_size,
                              hipStream_t stream) {
    (void)in_sizes; (void)n_in; (void)out_size; (void)ws_size;

    const float* edge_rep  = (const float*)d_in[0];
    const float* c5_rep    = (const float*)d_in[1];
    const float* c6_rep    = (const float*)d_in[2];
    const int*   edge_atoms = (const int*)d_in[3];
    const int*   c5_atoms   = (const int*)d_in[5];
    const int*   c6_atoms   = (const int*)d_in[7];
    const float* edge_w = (const float*)d_in[9];
    const float* edge_b = (const float*)d_in[10];
    const float* c5_w   = (const float*)d_in[11];
    const float* c5_b   = (const float*)d_in[12];
    const float* c6_w   = (const float*)d_in[13];
    const float* c6_b   = (const float*)d_in[14];
    const float* em_w1 = (const float*)d_in[15];
    const float* em_g1 = (const float*)d_in[16];
    const float* em_b1 = (const float*)d_in[17];
    const float* em_w2 = (const float*)d_in[18];
    const float* em_g2 = (const float*)d_in[19];
    const float* em_b2 = (const float*)d_in[20];
    const float* cm_w1 = (const float*)d_in[21];
    const float* cm_g1 = (const float*)d_in[22];
    const float* cm_b1 = (const float*)d_in[23];
    const float* cm_w2 = (const float*)d_in[24];
    const float* cm_g2 = (const float*)d_in[25];
    const float* cm_b2 = (const float*)d_in[26];
    const float* edge_eps  = (const float*)d_in[27];
    const float* cycle_eps = (const float*)d_in[28];

    float* out_edge = (float*)d_out;
    float* out_c5   = out_edge + (size_t)kERows * kH;
    float* out_c6   = out_c5   + (size_t)kC5Rows * kH;

    // ---- workspace carving (256B aligned regions) ----
    char*  base = (char*)d_ws;
    size_t off  = 0;
    auto take = [&](size_t bytes) -> char* {
        char* p = base + off;
        off += (bytes + 255) & ~(size_t)255;
        return p;
    };
    float*  ns1   = (float*)take((size_t)kNodes * kH  * 4);   // node sums of edge_rep
    float*  ns2   = (float*)take((size_t)kNodes * 256 * 4);   // node sums (reused as ns3)
    bf16_t* bigH  = (bf16_t*)take((size_t)kERows * 512 * 2);  // bf16 A mats / bf16 hidden
    float*  e2co5 = (float*)take((size_t)kC5Rows * kH * 4);
    float*  e2co6 = (float*)take((size_t)kC6Rows * kH * 4);
    bf16_t* xbf   = (bf16_t*)take((size_t)kERows * kH  * 2);  // MLP input (bf16)
    bf16_t* hnbf  = (bf16_t*)take((size_t)kERows * 256 * 2);  // BN1 output (bf16)
    bf16_t* wEt   = (bf16_t*)take((size_t)512 * kH * 2);
    bf16_t* w5t   = (bf16_t*)take((size_t)512 * kH * 2);
    bf16_t* w6t   = (bf16_t*)take((size_t)512 * kH * 2);
    bf16_t* em1t  = (bf16_t*)take((size_t)kH * 256 * 2);
    bf16_t* em2t  = (bf16_t*)take((size_t)256 * kH * 2);
    bf16_t* cm1t  = (bf16_t*)take((size_t)kH * 256 * 2);
    bf16_t* cm2t  = (bf16_t*)take((size_t)256 * kH * 2);
    float*  stats = (float*)take(512 * 4);

    // plain GEMM (no stats), wave = 16 x 128 output, grid.y = N/128
    auto gemm = [&](const bf16_t* Am, const bf16_t* Wt, const float* bias,
                    const float* res, const float* reps,
                    float* oF, bf16_t* oH, int M, int K, int N) {
        dim3 g((M / 16 + 7) / 8, N / 128);
        gemm_wmma_bf16<8, false><<<g, 256, 0, stream>>>(Am, Wt, bias, res, reps,
                                                        oF, oH, nullptr, M, K, N);
    };
    // GEMM with fused BN statistics
    auto gemm_st = [&](const bf16_t* Am, const bf16_t* Wt,
                       float* oF, bf16_t* oH, int M, int K, int N) {
        zero_f32<<<1, 256, 0, stream>>>(stats, 512);
        dim3 g((M / 16 + 7) / 8, N / 128);
        gemm_wmma_bf16<8, true><<<g, 256, 0, stream>>>(Am, Wt, nullptr, nullptr, nullptr,
                                                       oF, oH, stats, M, K, N);
    };

    // shared MLP: x(bf16)[M,128] -> Lin/BN/ReLU -> Lin/BN/ReLU -> out f32 [M,128]
    auto mlp_core = [&](const bf16_t* x_in, const bf16_t* w1t, const float* g1, const float* b1,
                        const bf16_t* w2t, const float* g2, const float* b2,
                        float* outp, int M) {
        gemm_st(x_in, w1t, nullptr, bigH, M, kH, 256);                 // h (bf16) + stats
        bn_apply_b2b<<<(M + 15) / 16, 256, 0, stream>>>(bigH, stats, g1, b1, 1.0f / M,
                                                        hnbf, M, 256);
        gemm_st(hnbf, w2t, outp, nullptr, M, 256, kH);                 // y (f32) + stats
        bn_apply_f32<<<(M + 15) / 16, kH, 0, stream>>>(outp, stats, g2, b2, 1.0f / M,
                                                       M, kH);
    };

    // ---- 0: weights -> transposed bf16 ----
    transpose_to_bf16<<<128, 256, 0, stream>>>(edge_w, wEt, 512, kH);
    transpose_to_bf16<<<128, 256, 0, stream>>>(c5_w,   w5t, 512, kH);
    transpose_to_bf16<<<128, 256, 0, stream>>>(c6_w,   w6t, 512, kH);
    transpose_to_bf16<<<128, 256, 0, stream>>>(em_w1,  em1t, kH, 256);
    transpose_to_bf16<<<128, 256, 0, stream>>>(em_w2,  em2t, 256, kH);
    transpose_to_bf16<<<128, 256, 0, stream>>>(cm_w1,  cm1t, kH, 256);
    transpose_to_bf16<<<128, 256, 0, stream>>>(cm_w2,  cm2t, 256, kH);

    // ---- 1: node sums of edge_rep ----
    zero_f32<<<2048, 256, 0, stream>>>(ns1, (size_t)kNodes * kH);
    scatter_rows<<<kERows, kH, 0, stream>>>(edge_rep, edge_atoms, ns1);

    // ---- 2: edge -> cycle5 (two nested _gather1 + linear) ----
    zero_f32<<<4096, 256, 0, stream>>>(ns2, (size_t)kNodes * 256);
    cyc_gather_scatter<5><<<kC5, kH, 0, stream>>>(ns1, c5_atoms, ns2);
    prep_e2c<5><<<kC5, 256, 0, stream>>>(ns2, c5_atoms, bigH);
    gemm(bigH, w5t, c5_b, nullptr, nullptr, e2co5, nullptr, kC5Rows, 512, kH);

    // ---- 3: edge -> cycle6 ----
    zero_f32<<<4096, 256, 0, stream>>>(ns2, (size_t)kNodes * 256);
    cyc_gather_scatter<6><<<kC6, kH, 0, stream>>>(ns1, c6_atoms, ns2);
    prep_e2c<6><<<kC6, 256, 0, stream>>>(ns2, c6_atoms, bigH);
    gemm(bigH, w6t, c6_b, nullptr, nullptr, e2co6, nullptr, kC6Rows, 512, kH);

    // ---- 4: cycles -> edge (node sums of both cats, then edge gather) ----
    zero_f32<<<4096, 256, 0, stream>>>(ns2, (size_t)kNodes * 256);   // ns3
    scatter_cat<<<kC5Rows, kH, 0, stream>>>(e2co5, c5_rep, c5_atoms, ns2);
    scatter_cat<<<kC6Rows, kH, 0, stream>>>(e2co6, c6_rep, c6_atoms, ns2);
    prep_edgeA<<<kEdges, 256, 0, stream>>>(ns2, edge_atoms, bigH);
    // x_edge = c2e @ W + b + (1+eps)*edge_rep  -> bf16 MLP input
    gemm(bigH, wEt, edge_b, edge_rep, edge_eps, nullptr, xbf, kERows, 512, kH);

    // ---- 5: edge MLP ----
    mlp_core(xbf, em1t, em_g1, em_b1, em2t, em_g2, em_b2, out_edge, kERows);

    // ---- 6: cycle MLPs (shared weights, per-tensor BN stats) ----
    add_scaled_bf16<<<4096, 256, 0, stream>>>(e2co5, c5_rep, cycle_eps, xbf,
                                              (size_t)kC5Rows * kH);
    mlp_core(xbf, cm1t, cm_g1, cm_b1, cm2t, cm_g2, cm_b2, out_c5, kC5Rows);

    add_scaled_bf16<<<4096, 256, 0, stream>>>(e2co6, c6_rep, cycle_eps, xbf,
                                              (size_t)kC6Rows * kH);
    mlp_core(xbf, cm1t, cm_g1, cm_b1, cm2t, cm_g2, cm_b2, out_c6, kC6Rows);
}
